// GraphSage_87754771791918
// MI455X (gfx1250) — compile-verified, matrix operand order checked by reading, not run on hardware
//
#include <hip/hip_runtime.h>
#include <hip/hip_bf16.h>
#include <cstddef>
#include <cstdint>

// ---------- types ----------
typedef __bf16 bf16;
typedef __attribute__((ext_vector_type(16))) __bf16 v16bf;
typedef __attribute__((ext_vector_type(8)))  __bf16 v8bf;
typedef __attribute__((ext_vector_type(8)))  float  v8f;

__device__ __forceinline__ bf16 f2bf(float x) { return (bf16)x; }
template <typename T>
__device__ __forceinline__ float to_f32(T x) { return (float)x; }

// ---------- small helpers ----------
// Build a 16x32 bf16 A/B fragment row for this lane from LDS.
// p points at element [row][ks + koff] where koff = (lane<16 ? 0 : 8).
// Elements 0..7  <- K = base+0..7 ; elements 8..15 <- K = base+16..23
__device__ __forceinline__ v16bf load_frag(const bf16* p) {
  v8bf lo = *(const v8bf*)(p);
  v8bf hi = *(const v8bf*)(p + 16);
  v16bf r;
#pragma unroll
  for (int i = 0; i < 8; ++i) { r[i] = lo[i]; r[i + 8] = hi[i]; }
  return r;
}

// ---------- kernels ----------

// Transpose + convert a small f32 weight [K,N] -> bf16 [N,K]
__global__ void transpose_w_kernel(const float* __restrict__ w,
                                   bf16* __restrict__ wt, int K, int N) {
  int id = blockIdx.x * blockDim.x + threadIdx.x;
  if (id >= K * N) return;
  int k = id / N, n = id % N;
  wt[(size_t)n * K + k] = f2bf(w[id]);
}

// Gather rows (optionally double-indirected) of a [*,128] matrix and store
// TRANSPOSED as bf16 [128][K] so GEMM B-loads are K-contiguous.
template <typename SrcT>
__global__ void gather_rows_T_kernel(const SrcT* __restrict__ src,
                                     const int* __restrict__ idx1,
                                     const int* __restrict__ idx2,
                                     int K, bf16* __restrict__ dstT) {
  __shared__ bf16 tile[128][130];   // 260B row (65 dwords, odd) -> conflict-free transpose
  int t  = threadIdx.x;             // 128 threads
  int k0 = blockIdx.x * 128;
  for (int kk = 0; kk < 128; ++kk) {
    int k = k0 + kk;
    float v = 0.f;
    if (k < K) {
      int row = idx1[k];
      if (idx2) row = idx2[row];
      v = to_f32(src[(size_t)row * 128 + t]);   // coalesced across t
    }
    tile[kk][t] = f2bf(v);
  }
  __syncthreads();
  int k = k0 + t;
  if (k < K) {
#pragma unroll 4
    for (int f = 0; f < 128; ++f)
      dstT[(size_t)f * K + k] = tile[t][f];     // coalesced across t
  }
}

// Gather rows of a [*,128] matrix into dst (row-major, stride ldd) as bf16.
template <typename SrcT>
__global__ void gather_rows_kernel(const SrcT* __restrict__ src,
                                   const int* __restrict__ idx1,
                                   const int* __restrict__ idx2,
                                   bf16* __restrict__ dst, int ldd, int nrows) {
  int d = blockIdx.x, t = threadIdx.x;   // 128 threads
  if (d >= nrows) return;
  int row = idx1[d];
  if (idx2) row = idx2[row];
  dst[(size_t)d * ldd + t] = f2bf(to_f32(src[(size_t)row * 128 + t]));
}

// C[M,128] = A[M,K] @ B[K,128], bf16 WMMA with f32 accumulation.
// A row-major (f32 or bf16), converted to bf16 while staging to LDS.
// Bt is B stored TRANSPOSED: Bt[n][k] -- staged to LDS by the Tensor Data
// Mover (tensor_load_to_lds, TENSORcnt), one issue per wave0 per chunk, with
// TDM LDS-padding matching the 72-element LDS row stride.
// Double-buffered LDS: chunk i+1's A loads + TDM issue overlap chunk i's WMMA.
template <typename AT>
__global__ void gemm_wmma_kernel(const AT* __restrict__ A, int M, int K,
                                 const bf16* __restrict__ Bt,
                                 bf16* __restrict__ C, int ldc, int relu) {
  __shared__ bf16 Al[2][32][72];    // 2 x (32 rows x 64 K, +8 pad)
  __shared__ bf16 Bl[2][128][72];   // 2 x (128 cols x 64 K, +8 pad)

  const int tid  = threadIdx.x;            // 256 threads = 8 waves
  const int wid  = tid >> 5;               // 0..7
  const int lane = tid & 31;
  const int msub = wid >> 2;               // 0..1  (M sub-tile of 16)
  const int nsub = wid & 3;                // 0..3  (N sub-tile of 32)
  const int ncol0 = nsub * 32;
  const int mt = blockIdx.x * 32;

  const int l15  = lane & 15;
  const int koff = (lane < 16) ? 0 : 8;

  v8f acc0 = {}, acc1 = {};

  // A-stage mapping: 256 threads -> 32 rows x 8 chunks of 8 elements
  const int ar = tid >> 3;                 // 0..31
  const int ac = (tid & 7) * 8;            // 0..56
  // B-stage mapping: 256 threads -> 128 n-rows x 2 chunks of 32 bf16 (tail only)
  const int bn  = tid >> 1;                // 0..127
  const int bk0 = (tid & 1) * 32;          // 0 or 32

  // Clamp the A row: rows >= M load row M-1; their accumulators are never stored.
  int gmA = mt + ar; if (gmA > M - 1) gmA = M - 1;
  const AT* aptr = &A[(size_t)gmA * K + ac];

  // ---- TDM descriptor constants (D# group1) ----
  // data_size=1 (2B), pad_enable, pad_interval=4 (32 dwords), pad_amount=3 (4 dwords)
  const unsigned g1w0 = (1u << 16) | (1u << 20) | (4u << 22) | (3u << 25);
  const unsigned g1w1 = ((unsigned)K & 0xFFFFu) << 16;                    // tensor_dim0 lo
  const unsigned g1w2 = (((unsigned)K >> 16) & 0xFFFFu) | (128u << 16);   // dim0 hi | tensor_dim1 lo
  const unsigned g1w3 = (64u << 16);                                      // tile_dim0 = 64
  const unsigned g1w4 = 128u;                                             // tile_dim1 = 128
  const unsigned g1w5 = (unsigned)K;                                      // tensor_dim0_stride lo32
  const unsigned ldsB[2] = { (unsigned)(uintptr_t)&Bl[0][0][0],
                             (unsigned)(uintptr_t)&Bl[1][0][0] };

  auto tdm_issue = [&](int buf, int kc) {
    unsigned long long ga = (unsigned long long)(uintptr_t)Bt + (unsigned long long)kc * 2ull;
    unsigned g0w0 = 1u;                                        // count=1, load descriptor
    unsigned g0w1 = ldsB[buf];                                 // lds_addr
    unsigned g0w2 = (unsigned)ga;                              // global_addr lo
    unsigned g0w3 = ((unsigned)(ga >> 32) & 0x01FFFFFFu) | 0x80000000u; // hi | type=2
    asm volatile(
        "s_mov_b32 s84, %0\n\t"
        "s_mov_b32 s85, %1\n\t"
        "s_mov_b32 s86, %2\n\t"
        "s_mov_b32 s87, %3\n\t"
        "s_mov_b32 s88, %4\n\t"
        "s_mov_b32 s89, %5\n\t"
        "s_mov_b32 s90, %6\n\t"
        "s_mov_b32 s91, %7\n\t"
        "s_mov_b32 s92, %8\n\t"
        "s_mov_b32 s93, %9\n\t"
        "s_mov_b32 s94, 0\n\t"
        "s_mov_b32 s95, 0\n\t"
        "tensor_load_to_lds s[84:87], s[88:95]"
        :: "s"(g0w0), "s"(g0w1), "s"(g0w2), "s"(g0w3),
           "s"(g1w0), "s"(g1w1), "s"(g1w2), "s"(g1w3),
           "s"(g1w4), "s"(g1w5)
        : "s84","s85","s86","s87","s88","s89","s90","s91",
          "s92","s93","s94","s95","memory");
  };

  auto stageA = [&](int buf, int kc) {
    struct V8 { AT v[8]; };
    V8 d = *(const V8*)(aptr + kc);
    v8bf ab;
#pragma unroll
    for (int i = 0; i < 8; ++i) ab[i] = f2bf(to_f32(d.v[i]));
    *(v8bf*)&Al[buf][ar][ac] = ab;
  };

  const int nch = K >> 6;   // full 64-wide chunks

  // ---- prologue: stage chunk 0 into buffer 0 ----
  if (nch > 0) {
    stageA(0, 0);
    if (wid == 0) tdm_issue(0, 0);
  }

  for (int i = 0; i < nch; ++i) {
    const int cur = i & 1;
    if (wid == 0) __builtin_amdgcn_s_wait_tensorcnt(0);  // chunk i's TDM done
    __syncthreads();                                     // publish buffer `cur`
    if (i + 1 < nch) {                                   // overlap chunk i+1 DMA
      stageA(cur ^ 1, (i + 1) << 6);
      if (wid == 0) tdm_issue(cur ^ 1, (i + 1) << 6);
    }
#pragma unroll
    for (int ks = 0; ks < 64; ks += 32) {
      v16bf a  = load_frag(&Al[cur][msub * 16 + l15][ks + koff]);
      v16bf b0 = load_frag(&Bl[cur][ncol0 + l15][ks + koff]);
      v16bf b1 = load_frag(&Bl[cur][ncol0 + 16 + l15][ks + koff]);
      acc0 = __builtin_amdgcn_wmma_f32_16x16x32_bf16(false, a, false, b0,
                                                     (short)0, acc0, false, false);
      acc1 = __builtin_amdgcn_wmma_f32_16x16x32_bf16(false, a, false, b1,
                                                     (short)0, acc1, false, false);
    }
  }

  // ---- tail chunk (K % 64, e.g. K=5000): checked, zero-padded, runs once ----
  const int kfull = nch << 6;
  if (kfull < K) {
    const int tb = nch & 1;    // free buffer
    v8bf ab;
#pragma unroll
    for (int i = 0; i < 8; ++i) {
      int gk = kfull + ac + i;
      ab[i] = f2bf((gk < K) ? to_f32(A[(size_t)gmA * K + gk]) : 0.f);
    }
    *(v8bf*)&Al[tb][ar][ac] = ab;
#pragma unroll
    for (int j = 0; j < 4; ++j) {
      v8bf bb;
#pragma unroll
      for (int i = 0; i < 8; ++i) {
        int gk = kfull + bk0 + j * 8 + i;
        bb[i] = (gk < K) ? Bt[(size_t)bn * K + gk] : f2bf(0.f);
      }
      *(v8bf*)&Bl[tb][bn][bk0 + j * 8] = bb;
    }
    __syncthreads();
#pragma unroll
    for (int ks = 0; ks < 64; ks += 32) {
      v16bf a  = load_frag(&Al[tb][msub * 16 + l15][ks + koff]);
      v16bf b0 = load_frag(&Bl[tb][ncol0 + l15][ks + koff]);
      v16bf b1 = load_frag(&Bl[tb][ncol0 + 16 + l15][ks + koff]);
      acc0 = __builtin_amdgcn_wmma_f32_16x16x32_bf16(false, a, false, b0,
                                                     (short)0, acc0, false, false);
      acc1 = __builtin_amdgcn_wmma_f32_16x16x32_bf16(false, a, false, b1,
                                                     (short)0, acc1, false, false);
    }
  }

  // ---- epilogue: C/D layout -> lane(0-15)=N, vgpr r: M=r (+8 for lanes 16-31)
  const int hiL = lane >> 4;
  const int mbase = mt + msub * 16 + hiL * 8;
#pragma unroll
  for (int r = 0; r < 8; ++r) {
    int m = mbase + r;
    if (m < M) {
      float v0 = acc0[r], v1 = acc1[r];
      if (relu) { v0 = fmaxf(v0, 0.f); v1 = fmaxf(v1, 0.f); }
      C[(size_t)m * ldc + ncol0 + l15]      = f2bf(v0);
      C[(size_t)m * ldc + ncol0 + 16 + l15] = f2bf(v1);
    }
  }
}

// logits = h2[row,:128] @ wcls[128,40]; softmax over 40 classes.
__global__ void classify_softmax_kernel(const bf16* __restrict__ h2,
                                        const float* __restrict__ wcls,
                                        float* __restrict__ out, int rows) {
  int row = blockIdx.x;
  if (row >= rows) return;
  __shared__ float hrow[128];
  __shared__ float logits[40];
  __shared__ float red[2];
  int t = threadIdx.x;  // 64 threads
  hrow[t]      = to_f32(h2[(size_t)row * 128 + t]);
  hrow[t + 64] = to_f32(h2[(size_t)row * 128 + t + 64]);
  __syncthreads();
  if (t < 40) {
    float acc = 0.f;
#pragma unroll 8
    for (int f = 0; f < 128; ++f) acc += hrow[f] * wcls[f * 40 + t];
    logits[t] = acc;
  }
  __syncthreads();
  if (t == 0) {
    float m = logits[0];
    for (int c = 1; c < 40; ++c) m = fmaxf(m, logits[c]);
    float s = 0.f;
    for (int c = 0; c < 40; ++c) s += __expf(logits[c] - m);
    red[0] = m; red[1] = s;
  }
  __syncthreads();
  if (t < 40) out[(size_t)row * 40 + t] = __expf(logits[t] - red[0]) / red[1];
}

// ---------- host ----------
extern "C" void kernel_launch(void* const* d_in, const int* in_sizes, int n_in,
                              void* d_out, int out_size, void* d_ws, size_t ws_size,
                              hipStream_t stream) {
  (void)in_sizes; (void)n_in; (void)out_size; (void)ws_size;

  const float* features  = (const float*)d_in[0];   // [100000,128]
  const int*   src_nodes = (const int*)  d_in[1];   // [60000]
  const int*   dst_idx_1 = (const int*)  d_in[2];   // [6000]
  const int*   src_idx_1 = (const int*)  d_in[3];   // [48000]
  const float* dif_mat_1 = (const float*)d_in[4];   // [6000,48000]
  const int*   dst_idx_0 = (const int*)  d_in[5];   // [1024]
  const int*   src_idx_0 = (const int*)  d_in[6];   // [5000]
  const float* dif_mat_0 = (const float*)d_in[7];   // [1024,5000]
  const float* w1        = (const float*)d_in[8];   // [256,128]
  const float* w2        = (const float*)d_in[9];   // [256,128]
  const float* wcls      = (const float*)d_in[10];  // [128,40]
  float* out = (float*)d_out;                       // [1024,40]

  constexpr int D1 = 6000, S1 = 48000, D0 = 1024, S0 = 5000;
  constexpr int H = 128, K2 = 256;

  char* ws = (char*)d_ws;
  size_t off = 0;
  auto alloc = [&](size_t bytes) -> void* {
    void* p = ws + off;
    off = (off + bytes + 255) & ~(size_t)255;
    return p;
  };
  bf16* B1t     = (bf16*)alloc((size_t)128 * S1 * sizeof(bf16)); // [128][48000]
  bf16* concat1 = (bf16*)alloc((size_t)D1 * 256 * sizeof(bf16)); // [6000][256]
  bf16* h1      = (bf16*)alloc((size_t)D1 * 128 * sizeof(bf16)); // [6000][128]
  bf16* src0t   = (bf16*)alloc((size_t)128 * S0 * sizeof(bf16)); // [128][5000]
  bf16* concat0 = (bf16*)alloc((size_t)D0 * 256 * sizeof(bf16)); // [1024][256]
  bf16* h2      = (bf16*)alloc((size_t)D0 * 128 * sizeof(bf16)); // [1024][128]
  bf16* wt1     = (bf16*)alloc((size_t)128 * K2 * sizeof(bf16)); // [128][256]
  bf16* wt2     = (bf16*)alloc((size_t)128 * K2 * sizeof(bf16)); // [128][256]

  // 1) tiny weight transposes
  transpose_w_kernel<<<(K2 * H + 255) / 256, 256, 0, stream>>>(w1, wt1, K2, H);
  transpose_w_kernel<<<(K2 * H + 255) / 256, 256, 0, stream>>>(w2, wt2, K2, H);

  // 2) level-1 gathers
  gather_rows_T_kernel<float><<<(S1 + 127) / 128, 128, 0, stream>>>(
      features, src_idx_1, src_nodes, S1, B1t);
  gather_rows_kernel<float><<<D1, 128, 0, stream>>>(
      features, dst_idx_1, src_nodes, concat1 + 128, 256, D1);

  // 3) diffusion GEMM 1: [6000,48000] @ [48000,128] -> concat1[:, :128]
  gemm_wmma_kernel<float><<<(D1 + 31) / 32, 256, 0, stream>>>(
      dif_mat_1, D1, S1, B1t, concat1, 256, 0);

  // 4) h1 = relu(concat1 @ w1)
  gemm_wmma_kernel<bf16><<<(D1 + 31) / 32, 256, 0, stream>>>(
      concat1, D1, K2, wt1, h1, 128, 1);

  // 5) level-2 gathers (from h1)
  gather_rows_T_kernel<bf16><<<(S0 + 127) / 128, 128, 0, stream>>>(
      h1, src_idx_0, nullptr, S0, src0t);
  gather_rows_kernel<bf16><<<D0, 128, 0, stream>>>(
      h1, dst_idx_0, nullptr, concat0 + 128, 256, D0);

  // 6) diffusion GEMM 2: [1024,5000] @ [5000,128] -> concat0[:, :128]
  gemm_wmma_kernel<float><<<(D0 + 31) / 32, 256, 0, stream>>>(
      dif_mat_0, D0, S0, src0t, concat0, 256, 0);

  // 7) h2 = relu(concat0 @ w2)
  gemm_wmma_kernel<bf16><<<(D0 + 31) / 32, 256, 0, stream>>>(
      concat0, D0, K2, wt2, h2, 128, 1);

  // 8) classifier + softmax
  classify_softmax_kernel<<<D0, 64, 0, stream>>>(h2, wcls, out, D0);
}
